// HeterogeneousAdaptiveSpectralGNN_34024730919243
// MI455X (gfx1250) — matrix operand order repaired and checked by reading, not compiled, not run on hardware
//
#include <hip/hip_runtime.h>
#include <stdint.h>
#include <stddef.h>

static constexpr int Hdim = 128;
static constexpr int NGr  = 256;
static constexpr int NCls = 10;

typedef __attribute__((ext_vector_type(16))) __bf16 v16bf;
typedef __attribute__((ext_vector_type(8)))  float  v8f;
typedef __attribute__((ext_vector_type(4)))  int    v4i;

#if defined(__HIP_DEVICE_COMPILE__) && __has_builtin(__builtin_amdgcn_global_load_async_to_lds_b128)
#define USE_ASYNC 1
#define ASYNC_WAIT() asm volatile("s_wait_asynccnt 0x0" ::: "memory")
#define AS_GLOBAL __attribute__((address_space(1)))
#define AS_LOCAL  __attribute__((address_space(3)))
#else
#define USE_ASYNC 0
#define ASYNC_WAIT()
#endif

// ---------------- utility kernels ----------------

__global__ void k_cvt_bf16(const float* __restrict__ in, __bf16* __restrict__ out, int n) {
  int i = blockIdx.x * blockDim.x + threadIdx.x;
  if (i < n) out[i] = (__bf16)in[i];
}

__global__ void k_edge_count(const int* __restrict__ dst, int E, float* __restrict__ cnt) {
  int e = blockIdx.x * blockDim.x + threadIdx.x;
  if (e < E) atomicAdd(&cnt[dst[e]], 1.0f);
}

__global__ void k_invert(float* __restrict__ c, int n) {
  int i = blockIdx.x * blockDim.x + threadIdx.x;
  if (i < n) c[i] = 1.0f / fmaxf(c[i], 1.0f);
}

// one wave per edge: lane l moves 4 consecutive features (128 = 32 lanes * 4)
__global__ void k_edge_scatter(const int* __restrict__ src, const int* __restrict__ dst, int E,
                               const float* __restrict__ h, float* __restrict__ agg) {
  long long t = (long long)blockIdx.x * blockDim.x + threadIdx.x;
  int e = (int)(t >> 5);
  if (e >= E) return;
  int lane = (int)(t & 31);
  int s = src[e], d = dst[e];
  float4 v = ((const float4*)(h + (size_t)s * Hdim))[lane];
  float* ap = agg + (size_t)d * Hdim + lane * 4;
  atomicAdd(ap + 0, v.x); atomicAdd(ap + 1, v.y);
  atomicAdd(ap + 2, v.z); atomicAdd(ap + 3, v.w);
}

__global__ void k_pool_scatter(const float* __restrict__ h, const int* __restrict__ batch, int N,
                               float* __restrict__ pool, float* __restrict__ cnt) {
  long long t = (long long)blockIdx.x * blockDim.x + threadIdx.x;
  int node = (int)(t >> 5);
  if (node >= N) return;
  int lane = (int)(t & 31);
  int g = batch[node];
  float4 v = ((const float4*)(h + (size_t)node * Hdim))[lane];
  float* pp = pool + (size_t)g * Hdim + lane * 4;
  atomicAdd(pp + 0, v.x); atomicAdd(pp + 1, v.y);
  atomicAdd(pp + 2, v.z); atomicAdd(pp + 3, v.w);
  if (lane == 0) atomicAdd(&cnt[g], 1.0f);
}

// fold lin bias + eval BatchNorm into per-feature (scale, shift)
__global__ void k_bn_prep(const float* __restrict__ bl0, const float* __restrict__ bl1,
                          const float* __restrict__ g,  const float* __restrict__ b,
                          const float* __restrict__ rm, const float* __restrict__ rv,
                          float* __restrict__ scaleE, float* __restrict__ shiftE) {
  int j = threadIdx.x;
  if (j >= Hdim) return;
  float k = rsqrtf(rv[j] + 1e-5f) * g[j];
  float bias = bl0[j] + (bl1 ? bl1[j] : 0.0f);
  scaleE[j] = k;
  shiftE[j] = (bias - rm[j]) * k + b[j];
}

// ---------------- WMMA GEMM with LDS-staged weights ----------------
// out[m,:] = relu(scaleE*(sum_c rowScale_c[m]*A_c[m,:]@W_c^T)+shiftE)

template <int K>
__device__ __forceinline__ void gemm_contrib_lds(const float* __restrict__ A,
                                                 const __bf16* __restrict__ sWb,
                                                 const float* __restrict__ rowScale,
                                                 int arow, int lane, v8f acc[8]) {
  const int rloc = lane & 15;
  const int khA  = (lane >> 4) << 4;   // A frag: lanes16-31 cover K 16..31 (contiguous)
  const int khB  = (lane >> 4) << 4;   // B frag: lanes16-31 cover K 16..31 (contiguous)
  constexpr int P = K + 16;            // LDS pitch in bf16 (288B rows for K=128: bank-spread)
  const float* Arow = A + (size_t)arow * K;
  const float  s = rowScale ? rowScale[arow] : 1.0f;
  #pragma unroll
  for (int kk = 0; kk < K; kk += 32) {
    const float4* p = (const float4*)(Arow + kk + khA);
    float4 x0 = p[0], x1 = p[1];
    const float4* q = (const float4*)(Arow + kk + khA + 8);
    float4 y0 = q[0], y1 = q[1];
    v16bf a;
    a[0]  = (__bf16)(x0.x * s); a[1]  = (__bf16)(x0.y * s);
    a[2]  = (__bf16)(x0.z * s); a[3]  = (__bf16)(x0.w * s);
    a[4]  = (__bf16)(x1.x * s); a[5]  = (__bf16)(x1.y * s);
    a[6]  = (__bf16)(x1.z * s); a[7]  = (__bf16)(x1.w * s);
    a[8]  = (__bf16)(y0.x * s); a[9]  = (__bf16)(y0.y * s);
    a[10] = (__bf16)(y0.z * s); a[11] = (__bf16)(y0.w * s);
    a[12] = (__bf16)(y1.x * s); a[13] = (__bf16)(y1.y * s);
    a[14] = (__bf16)(y1.z * s); a[15] = (__bf16)(y1.w * s);
    #pragma unroll
    for (int n = 0; n < 8; ++n) {
      // B[k][col] = W[col][k]; read 32B per lane from padded LDS image
      const __bf16* Wp = sWb + (size_t)(n * 16 + rloc) * P + kk + khB;
      v16bf b = *(const v16bf*)Wp;
      acc[n] = __builtin_amdgcn_wmma_f32_16x16x32_bf16(false, a, false, b,
                                                       (short)0, acc[n], false, false);
    }
  }
}

__global__ void __launch_bounds__(256) k_sage_gemm(
    const float* A0, const __bf16* W0, const float* S0, int K0,
    const float* A1, const __bf16* W1, const float* S1, int K1,
    const float* A2, const __bf16* W2, const float* S2, int K2,
    const float* A3, const __bf16* W3, const float* S3, int K3,
    int nC, int nRows,
    const float* __restrict__ scaleE, const float* __restrict__ shiftE,
    float* __restrict__ out)
{
  __shared__ __bf16 sW[2][Hdim * (Hdim + 16)];   // double-buffered padded weight image

  const int wave = threadIdx.x >> 5;
  const int lane = threadIdx.x & 31;
  const int row0 = (blockIdx.x * 8 + wave) * 16;
  const bool active = (row0 < nRows);            // wave-uniform; inactive waves still stage+barrier

  int arow = row0 + (lane & 15);
  if (arow >= nRows) arow = nRows - 1;           // clamp keeps EXEC all-ones

  // cooperative stage of one weight matrix into padded LDS buffer (async DMA when available)
  auto stage = [&](const __bf16* W, int K, __bf16* sbuf) {
    const int P = K + 16;
    const int lsh = (K >= 128) ? 4 : (K >= 64 ? 3 : 2);   // log2(K/8)
    const int msk = (1 << lsh) - 1;
    const int total = Hdim << lsh;                        // 16B (8 bf16) chunks
    for (int i = threadIdx.x; i < total; i += 256) {
      int r = i >> lsh, cv = i & msk;
      const __bf16* gp = W + (size_t)r * K + cv * 8;
      __bf16* sp = sbuf + r * P + cv * 8;
#if USE_ASYNC
      __builtin_amdgcn_global_load_async_to_lds_b128(
          (AS_GLOBAL v4i*)(AS_GLOBAL const void*)(const void*)gp,
          (AS_LOCAL  v4i*)(AS_LOCAL void*)(void*)sp,
          0, 0);
#else
      *(uint4*)sp = *(const uint4*)gp;
#endif
    }
  };

  v8f acc[8] = {};

  // uniform-K dispatch to fully-unrolled WMMA bodies
  auto contrib = [&](const float* A, const float* S, int K, const __bf16* sbuf) {
    if (K == 128)     gemm_contrib_lds<128>(A, sbuf, S, arow, lane, acc);
    else if (K == 64) gemm_contrib_lds<64>(A, sbuf, S, arow, lane, acc);
    else              gemm_contrib_lds<32>(A, sbuf, S, arow, lane, acc);
  };

  // prologue: stage contrib 0
  stage(W0, K0, sW[0]);
  ASYNC_WAIT();
  __syncthreads();

  for (int c = 0; c < nC; ++c) {
    const float* A = A0; const float* S = S0; int K = K0;
    if (c == 1)      { A = A1; S = S1; K = K1; }
    else if (c == 2) { A = A2; S = S2; K = K2; }
    else if (c == 3) { A = A3; S = S3; K = K3; }

    if (c + 1 < nC) {            // kick off next weight DMA while computing this one
      const __bf16* Wn = W1; int Kn = K1;
      if (c + 1 == 2)      { Wn = W2; Kn = K2; }
      else if (c + 1 == 3) { Wn = W3; Kn = K3; }
      stage(Wn, Kn, sW[(c + 1) & 1]);
    }
    if (active) contrib(A, S, K, sW[c & 1]);
    if (c + 1 < nC) {
      ASYNC_WAIT();
      __syncthreads();
    }
  }

  if (!active) return;

  // C layout: lane%16 -> column in tile, vgpr r -> row r (+8 for upper half-wave)
  const int colb = lane & 15;
  const int roff = (lane >> 4) << 3;
  #pragma unroll
  for (int n = 0; n < 8; ++n) {
    int col = n * 16 + colb;
    float se = scaleE ? scaleE[col] : 1.0f;
    float sh = shiftE[col];
    #pragma unroll
    for (int r = 0; r < 8; ++r) {
      int row = row0 + roff + r;
      if (row < nRows) {
        float v = acc[n][r] * se + sh;
        out[(size_t)row * Hdim + col] = fmaxf(v, 0.0f);
      }
    }
  }
}

// ---------------- pooling MLP head (tiny: 256x128 -> 64 -> 10) ----------------

__global__ void __launch_bounds__(128) k_head(
    const float* __restrict__ pa, const float* __restrict__ pm,
    const float* __restrict__ ca, const float* __restrict__ cm,
    const float* __restrict__ W1, const float* __restrict__ b1,
    const float* __restrict__ W2, const float* __restrict__ b2,
    float* __restrict__ out)
{
  __shared__ float gsh[Hdim];
  __shared__ float h1[64];
  int gid = blockIdx.x;
  int t = threadIdx.x;
  float inva = 1.0f / fmaxf(ca[gid], 1.0f);
  float invm = 1.0f / fmaxf(cm[gid], 1.0f);
  gsh[t] = (pa[(size_t)gid * Hdim + t] * inva + pm[(size_t)gid * Hdim + t] * invm) * 0.5f;
  __syncthreads();
  if (t < 64) {
    float accv = b1[t];
    #pragma unroll 4
    for (int k = 0; k < Hdim; ++k) accv += gsh[k] * W1[t * Hdim + k];
    h1[t] = fmaxf(accv, 0.0f);
  }
  __syncthreads();
  if (t < NCls) {
    float accv = b2[t];
    #pragma unroll 4
    for (int k = 0; k < 64; ++k) accv += h1[k] * W2[t * 64 + k];
    out[gid * NCls + t] = accv;
  }
  if (gid == 0 && t == 0) out[NGr * NCls] = 0.0f;   // trailing aux scalar
}

// ---------------- host launch ----------------

extern "C" void kernel_launch(void* const* d_in, const int* in_sizes, int n_in,
                              void* d_out, int out_size, void* d_ws, size_t ws_size,
                              hipStream_t stream)
{
  (void)n_in; (void)ws_size; (void)out_size;
  // inputs flattened in jax pytree (sorted-key) order
  const float* x_atom = (const float*)d_in[0];
  const float* x_mot  = (const float*)d_in[1];
  const int* ei_aa = (const int*)d_in[2];
  const int* ei_am = (const int*)d_in[3];
  const int* ei_ma = (const int*)d_in[4];
  const int* batch_atom = (const int*)d_in[5];
  const int* batch_mot  = (const int*)d_in[6];
  const float* fc1W = (const float*)d_in[7];
  const float* fc1b = (const float*)d_in[8];
  const float* fc2W = (const float*)d_in[9];
  const float* fc2b = (const float*)d_in[10];
  const int NLAY = 4;
  // layer dict sorted: aa(Wl,Wr,bl) am(Wl,Wr,bl) bn_atom(b,g,rm,rv) bn_motif(b,g,rm,rv) ma(Wl,Wr,bl)
  auto LP = [&](int l, int off) { return (const float*)d_in[11 + 17 * l + off]; };
  const int projBase = 11 + 17 * NLAY;
  const float* projAW = (const float*)d_in[projBase + 0];
  const float* projAb = (const float*)d_in[projBase + 1];
  const float* projMW = (const float*)d_in[projBase + 2];
  const float* projMb = (const float*)d_in[projBase + 3];

  const int EAA = in_sizes[2] / 2;
  const int EAM = in_sizes[3] / 2;
  const int EMA = in_sizes[4] / 2;
  const int NA  = in_sizes[5];
  const int NM  = in_sizes[6];
  const int KA  = in_sizes[0] / NA;   // 32
  const int KM  = in_sizes[1] / NM;   // 64

  // workspace carve (256B aligned)
  char* wp = (char*)d_ws;
  auto carve = [&](size_t bytes) { void* r = (void*)wp; wp += (bytes + 255) & ~(size_t)255; return r; };
  float* haA = (float*)carve((size_t)NA * Hdim * 4);
  float* haB = (float*)carve((size_t)NA * Hdim * 4);
  float* hmA = (float*)carve((size_t)NM * Hdim * 4);
  float* hmB = (float*)carve((size_t)NM * Hdim * 4);
  float* agg_aa = (float*)carve((size_t)NA * Hdim * 4);
  float* agg_ma = (float*)carve((size_t)NA * Hdim * 4);
  float* agg_am = (float*)carve((size_t)NM * Hdim * 4);
  float* inv_aa = (float*)carve((size_t)NA * 4);
  float* inv_ma = (float*)carve((size_t)NA * 4);
  float* inv_am = (float*)carve((size_t)NM * 4);
  float* poolA  = (float*)carve((size_t)NGr * Hdim * 4);
  float* poolM  = (float*)carve((size_t)NGr * Hdim * 4);
  float* cga    = (float*)carve((size_t)NGr * 4);
  float* cgm    = (float*)carve((size_t)NGr * 4);
  __bf16* projAWb = (__bf16*)carve((size_t)Hdim * KA * 2);
  __bf16* projMWb = (__bf16*)carve((size_t)Hdim * KM * 2);
  __bf16* wb[4][6];
  for (int l = 0; l < NLAY; ++l)
    for (int m = 0; m < 6; ++m)
      wb[l][m] = (__bf16*)carve((size_t)Hdim * Hdim * 2);
  float *scA[4], *shA[4], *scM[4], *shM[4];
  for (int l = 0; l < NLAY; ++l) {
    scA[l] = (float*)carve(Hdim * 4); shA[l] = (float*)carve(Hdim * 4);
    scM[l] = (float*)carve(Hdim * 4); shM[l] = (float*)carve(Hdim * 4);
  }

  auto cdiv = [](long long a, long long b) { return (int)((a + b - 1) / b); };

  // weights -> bf16 (tiny; stay hot in L2)
  k_cvt_bf16<<<cdiv(Hdim*KA,256),256,0,stream>>>(projAW, projAWb, Hdim*KA);
  k_cvt_bf16<<<cdiv(Hdim*KM,256),256,0,stream>>>(projMW, projMWb, Hdim*KM);
  const int wOff[6] = {0, 1, 3, 4, 14, 15};  // aaWl,aaWr,amWl,amWr,maWl,maWr
  for (int l = 0; l < NLAY; ++l)
    for (int m = 0; m < 6; ++m)
      k_cvt_bf16<<<cdiv(Hdim*Hdim,256),256,0,stream>>>(LP(l, wOff[m]), wb[l][m], Hdim*Hdim);

  // degree counts (edges static -> compute once, reuse across layers)
  (void)hipMemsetAsync(inv_aa, 0, (size_t)NA*4, stream);
  (void)hipMemsetAsync(inv_ma, 0, (size_t)NA*4, stream);
  (void)hipMemsetAsync(inv_am, 0, (size_t)NM*4, stream);
  k_edge_count<<<cdiv(EAA,256),256,0,stream>>>(ei_aa + EAA, EAA, inv_aa);
  k_edge_count<<<cdiv(EMA,256),256,0,stream>>>(ei_ma + EMA, EMA, inv_ma);
  k_edge_count<<<cdiv(EAM,256),256,0,stream>>>(ei_am + EAM, EAM, inv_am);
  k_invert<<<cdiv(NA,256),256,0,stream>>>(inv_aa, NA);
  k_invert<<<cdiv(NA,256),256,0,stream>>>(inv_ma, NA);
  k_invert<<<cdiv(NM,256),256,0,stream>>>(inv_am, NM);

  // input projections (WMMA, bias+ReLU epilogue via shiftE)
  k_sage_gemm<<<cdiv((NA+15)/16,8),256,0,stream>>>(
      x_atom, projAWb, nullptr, KA,
      nullptr, nullptr, nullptr, 0, nullptr, nullptr, nullptr, 0, nullptr, nullptr, nullptr, 0,
      1, NA, nullptr, projAb, haA);
  k_sage_gemm<<<cdiv((NM+15)/16,8),256,0,stream>>>(
      x_mot, projMWb, nullptr, KM,
      nullptr, nullptr, nullptr, 0, nullptr, nullptr, nullptr, 0, nullptr, nullptr, nullptr, 0,
      1, NM, nullptr, projMb, hmA);

  float* ha_cur = haA; float* ha_nxt = haB;
  float* hm_cur = hmA; float* hm_nxt = hmB;

  for (int l = 0; l < NLAY; ++l) {
    // fused bias+BN epilogue coefficients
    k_bn_prep<<<1,128,0,stream>>>(LP(l,2), LP(l,16), LP(l,7), LP(l,6), LP(l,8), LP(l,9), scA[l], shA[l]);
    k_bn_prep<<<1,128,0,stream>>>(LP(l,5), nullptr,  LP(l,11), LP(l,10), LP(l,12), LP(l,13), scM[l], shM[l]);

    (void)hipMemsetAsync(agg_aa, 0, (size_t)NA*Hdim*4, stream);
    (void)hipMemsetAsync(agg_ma, 0, (size_t)NA*Hdim*4, stream);
    (void)hipMemsetAsync(agg_am, 0, (size_t)NM*Hdim*4, stream);

    k_edge_scatter<<<cdiv((long long)EAA*32,256),256,0,stream>>>(ei_aa, ei_aa+EAA, EAA, ha_cur, agg_aa);
    k_edge_scatter<<<cdiv((long long)EMA*32,256),256,0,stream>>>(ei_ma, ei_ma+EMA, EMA, hm_cur, agg_ma);
    k_edge_scatter<<<cdiv((long long)EAM*32,256),256,0,stream>>>(ei_am, ei_am+EAM, EAM, ha_cur, agg_am);

    // atoms: mean_aa@Wl_aa + ha@Wr_aa + mean_ma@Wl_ma + ha@Wr_ma -> BN -> ReLU
    k_sage_gemm<<<cdiv((NA+15)/16,8),256,0,stream>>>(
        agg_aa, wb[l][0], inv_aa, Hdim,
        ha_cur, wb[l][1], nullptr, Hdim,
        agg_ma, wb[l][4], inv_ma, Hdim,
        ha_cur, wb[l][5], nullptr, Hdim,
        4, NA, scA[l], shA[l], ha_nxt);
    // motifs: mean_am@Wl_am + hm@Wr_am -> BN -> ReLU
    k_sage_gemm<<<cdiv((NM+15)/16,8),256,0,stream>>>(
        agg_am, wb[l][2], inv_am, Hdim,
        hm_cur, wb[l][3], nullptr, Hdim,
        nullptr, nullptr, nullptr, 0, nullptr, nullptr, nullptr, 0,
        2, NM, scM[l], shM[l], hm_nxt);

    float* t0 = ha_cur; ha_cur = ha_nxt; ha_nxt = t0;
    float* t1 = hm_cur; hm_cur = hm_nxt; hm_nxt = t1;
  }

  // pooling + head
  (void)hipMemsetAsync(poolA, 0, (size_t)NGr*Hdim*4, stream);
  (void)hipMemsetAsync(poolM, 0, (size_t)NGr*Hdim*4, stream);
  (void)hipMemsetAsync(cga, 0, (size_t)NGr*4, stream);
  (void)hipMemsetAsync(cgm, 0, (size_t)NGr*4, stream);
  k_pool_scatter<<<cdiv((long long)NA*32,256),256,0,stream>>>(ha_cur, batch_atom, NA, poolA, cga);
  k_pool_scatter<<<cdiv((long long)NM*32,256),256,0,stream>>>(hm_cur, batch_mot, NM, poolM, cgm);
  k_head<<<NGr,128,0,stream>>>(poolA, poolM, cga, cgm, fc1W, fc1b, fc2W, fc2b, (float*)d_out);
}
